// KMeans_60060822667905
// MI455X (gfx1250) — matrix operand order repaired and checked by reading, not compiled, not run on hardware
//
#include <hip/hip_runtime.h>
#include <hip/hip_bf16.h>

// ---------------- problem constants (from reference) ----------------
#define B_DIM   32
#define C_DIM   256
#define T_DIM   1024
#define N_PTS   (B_DIM * T_DIM)     // 32768 points
#define K_CTR   1024                // centers
#define NTILES  (K_CTR / 16)        // 64 column tiles
#define EMA_DEC 0.99f
#define EPS_    1e-5f
#define ITERS   10

typedef __attribute__((ext_vector_type(16))) __bf16 v16bf;
typedef __attribute__((ext_vector_type(8)))  __bf16 v8bf;
typedef __attribute__((ext_vector_type(8)))  float  v8f;
typedef __attribute__((vector_size(16)))     int    vi4;   // builtin's pointee type

union AFrag { v16bf v; struct { v8bf lo, hi; } s; };

#if __has_builtin(__builtin_amdgcn_global_load_async_to_lds_b128) && \
    __has_builtin(__builtin_amdgcn_s_wait_asynccnt)
#define HAVE_ASYNC 1
#else
#define HAVE_ASYNC 0
#endif

__device__ __forceinline__ unsigned short f32_to_bf16_rne(float f) {
  union { float f; unsigned u; } v; v.f = f;
  unsigned u = v.u;
  u += 0x7fffu + ((u >> 16) & 1u);   // round-to-nearest-even
  return (unsigned short)(u >> 16);
}

// ---------------------------------------------------------------
// 1) x [b, c, t] f32  ->  xf_bf16 [n=b*T+t, c]  (coalesced reads)
// ---------------------------------------------------------------
__global__ void prep_bf16(const float* __restrict__ x,
                          unsigned short* __restrict__ xfbf) {
  int tid = blockIdx.x * blockDim.x + threadIdx.x;     // over B*C*T
  float v = x[tid];
  int t = tid & (T_DIM - 1);
  int c = (tid >> 10) & (C_DIM - 1);
  int b = tid >> 18;
  xfbf[((b << 10) + t) * C_DIM + c] = f32_to_bf16_rne(v);
}

// ---------------------------------------------------------------
// 2) ema0[k] = xf[perm(k)]  (deterministic bijection on Z_2^15)
// ---------------------------------------------------------------
__global__ void init_centers(const float* __restrict__ x,
                             float* __restrict__ ema) {
  int k = blockIdx.x, c = threadIdx.x;
  int n = (k * 20021) & (N_PTS - 1);   // odd multiplier -> permutation
  int b = n >> 10, t = n & (T_DIM - 1);
  ema[k * C_DIM + c] = x[(b * C_DIM + c) * T_DIM + t];
}

// ---------------------------------------------------------------
// 3) optional EMA update, then pack centers to bf16 + c_sq[k]
// ---------------------------------------------------------------
__global__ void update_pack(float* __restrict__ ema,
                            const float* __restrict__ sums,
                            const float* __restrict__ counts,
                            unsigned short* __restrict__ cbf,
                            float* __restrict__ c_sq,
                            int do_update) {
  __shared__ float red[C_DIM];
  int k = blockIdx.x, c = threadIdx.x;
  float e = ema[k * C_DIM + c];
  if (do_update) {
    float cnt = counts[k];
    float ctr = (cnt > 0.0f) ? (sums[k * C_DIM + c] / (cnt + EPS_)) : e;
    e = e * EMA_DEC + (1.0f - EMA_DEC) * ctr;
    ema[k * C_DIM + c] = e;
  }
  cbf[k * C_DIM + c] = f32_to_bf16_rne(e);
  red[c] = e * e;
  __syncthreads();
  for (int s = C_DIM / 2; s > 0; s >>= 1) {
    if (c < s) red[c] += red[c + s];
    __syncthreads();
  }
  if (c == 0) c_sq[k] = red[0];
}

// ---------------------------------------------------------------
// 4) zero sums+counts (contiguous)
// ---------------------------------------------------------------
__global__ void zero_buf(float* __restrict__ p, int n) {
  int i = blockIdx.x * blockDim.x + threadIdx.x;
  if (i < n) p[i] = 0.0f;
}

// ---------------------------------------------------------------
// 5) segment sums via fp32 L2 atomics (block = one point)
// ---------------------------------------------------------------
__global__ void accumulate(const float* __restrict__ x,
                           const int* __restrict__ labels,
                           float* __restrict__ sums,
                           float* __restrict__ counts) {
  int n = blockIdx.x, c = threadIdx.x;
  int lab = labels[n];
  int b = n >> 10, t = n & (T_DIM - 1);
  float v = x[(b * C_DIM + c) * T_DIM + t];
  unsafeAtomicAdd(&sums[lab * C_DIM + c], v);
  if (c == 0) unsafeAtomicAdd(&counts[lab], 1.0f);
}

// ---------------------------------------------------------------
// Stage one B tile (16 centers x 256 feats bf16 = 8 KB) into LDS in
// fragment-linear order:  LDS[(m*16 + l16)*16 .. +15] =
//                         cbf[(nt*16 + l16)*256 + m*16 .. +15]
// Each of 256 threads moves one 32 B chunk (two async B128 ops).
// ---------------------------------------------------------------
__device__ __forceinline__ void stage_tile(const unsigned short* __restrict__ cbf,
                                           unsigned short* dst_lds,
                                           int nt, int tid) {
  int m = tid >> 4, l16 = tid & 15;
  unsigned short* src =
      const_cast<unsigned short*>(cbf) + (size_t)(nt * 16 + l16) * C_DIM + m * 16;
  unsigned short* dst = dst_lds + (m * 16 + l16) * 16;
#if HAVE_ASYNC
  __builtin_amdgcn_global_load_async_to_lds_b128(
      (__attribute__((address_space(1))) vi4*)src,
      (__attribute__((address_space(3))) vi4*)dst, 0, 0);
  __builtin_amdgcn_global_load_async_to_lds_b128(
      (__attribute__((address_space(1))) vi4*)(src + 8),
      (__attribute__((address_space(3))) vi4*)(dst + 8), 0, 0);
#else
  *(uint4*)dst = *(const uint4*)src;
  *(uint4*)(dst + 8) = *(const uint4*)(src + 8);
#endif
}

// ---------------------------------------------------------------
// 6) fused distance-GEMM + argmin.
//    Block = 256 threads = 8 waves; wave owns 2 M-groups (32 rows),
//    A (K=256) pinned in VGPRs.  B tiles double-buffered in LDS via
//    async loads, each staged fragment feeds 2 WMMAs.
//    Rank by c_sq[k] - 2*dot (x^2 row-constant dropped).
// ---------------------------------------------------------------
__global__ __launch_bounds__(256)
void assign_wmma(const unsigned short* __restrict__ xfbf,
                 const unsigned short* __restrict__ cbf,
                 const float* __restrict__ c_sq,
                 int* __restrict__ labels,
                 float* __restrict__ out_f) {
  __shared__ unsigned short btile[2][16 * 16 * 16];   // 2 x 8 KB

  const int tid     = threadIdx.x;
  const int lane    = tid & 31;
  const int wave    = tid >> 5;
  const int rowBase = blockIdx.x * 256 + wave * 32;   // 2 groups of 16 rows
  const int half    = lane >> 4;                      // 0: lanes 0-15, 1: 16-31
  const int l16     = lane & 15;

  // A fragments (contiguous layout: lane<16 holds K 0-15, lane>=16 K 16-31)
  AFrag a0[8], a1[8];
  const unsigned short* ar0 = xfbf + (size_t)(rowBase + l16) * C_DIM;
  const unsigned short* ar1 = xfbf + (size_t)(rowBase + 16 + l16) * C_DIM;
#pragma unroll
  for (int ks = 0; ks < 8; ++ks) {
    a0[ks].s.lo = *(const v8bf*)(ar0 + ks * 32 + half * 16);
    a0[ks].s.hi = *(const v8bf*)(ar0 + ks * 32 + half * 16 + 8);
    a1[ks].s.lo = *(const v8bf*)(ar1 + ks * 32 + half * 16);
    a1[ks].s.hi = *(const v8bf*)(ar1 + ks * 32 + half * 16 + 8);
  }

  float best0[8], best1[8];
  int   bidx0[8], bidx1[8];
#pragma unroll
  for (int v = 0; v < 8; ++v) {
    best0[v] = 3.4e38f; bidx0[v] = 0;
    best1[v] = 3.4e38f; bidx1[v] = 0;
  }

  stage_tile(cbf, &btile[0][0], 0, tid);              // prologue: tile 0

  for (int nt = 0; nt < NTILES; ++nt) {
    const int cur = nt & 1;
    if (nt < NTILES - 1) {
      stage_tile(cbf, &btile[cur ^ 1][0], nt + 1, tid);
#if HAVE_ASYNC
      __builtin_amdgcn_s_wait_asynccnt(2);   // tile nt landed (in-order), nt+1 in flight
#endif
    } else {
#if HAVE_ASYNC
      __builtin_amdgcn_s_wait_asynccnt(0);   // last tile fully landed
#endif
    }
    __syncthreads();                         // LDS writes visible to all waves

    const unsigned short* bt = &btile[cur][0];
    v8f acc0 = {}, acc1 = {};
#pragma unroll
    for (int ks = 0; ks < 8; ++ks) {
      // B layout (32x16 bf16): lanes 0-15 hold K=0-15, lanes 16-31 K=16-31
      v16bf bfrag = *(const v16bf*)(bt + (((ks * 2 + half) * 16) + l16) * 16);
      acc0 = __builtin_amdgcn_wmma_f32_16x16x32_bf16(
          false, a0[ks].v, false, bfrag, (short)0, acc0, false, false);
      acc1 = __builtin_amdgcn_wmma_f32_16x16x32_bf16(
          false, a1[ks].v, false, bfrag, (short)0, acc1, false, false);
    }

    const int ncol = nt * 16 + l16;                   // this lane's column
    const float cs = c_sq[ncol];
#pragma unroll
    for (int v = 0; v < 8; ++v) {                     // acc[v]: row v + half*8
      float d0 = cs - 2.0f * acc0[v];
      float d1 = cs - 2.0f * acc1[v];
      if (d0 < best0[v]) { best0[v] = d0; bidx0[v] = ncol; }
      if (d1 < best1[v]) { best1[v] = d1; bidx1[v] = ncol; }
    }
    __syncthreads();                                  // done reading btile[cur]
  }

  // min across the 16 lanes of each half (ties -> lowest index, like argmin)
#pragma unroll
  for (int v = 0; v < 8; ++v) {
    float bv0 = best0[v]; int bi0 = bidx0[v];
    float bv1 = best1[v]; int bi1 = bidx1[v];
#pragma unroll
    for (int off = 1; off < 16; off <<= 1) {
      float ov0 = __shfl_xor(bv0, off, 32);
      int   oi0 = __shfl_xor(bi0, off, 32);
      float ov1 = __shfl_xor(bv1, off, 32);
      int   oi1 = __shfl_xor(bi1, off, 32);
      if (ov0 < bv0 || (ov0 == bv0 && oi0 < bi0)) { bv0 = ov0; bi0 = oi0; }
      if (ov1 < bv1 || (ov1 == bv1 && oi1 < bi1)) { bv1 = ov1; bi1 = oi1; }
    }
    best0[v] = bv0; bidx0[v] = bi0;
    best1[v] = bv1; bidx1[v] = bi1;
  }

  if (l16 == 0) {
#pragma unroll
    for (int v = 0; v < 8; ++v) {
      int r0 = rowBase + half * 8 + v;
      int r1 = rowBase + 16 + half * 8 + v;
      labels[r0] = bidx0[v];
      labels[r1] = bidx1[v];
      if (out_f) { out_f[r0] = (float)bidx0[v]; out_f[r1] = (float)bidx1[v]; }
    }
  }
}

// ---------------------------------------------------------------
extern "C" void kernel_launch(void* const* d_in, const int* in_sizes, int n_in,
                              void* d_out, int out_size, void* d_ws, size_t ws_size,
                              hipStream_t stream) {
  (void)in_sizes; (void)n_in; (void)out_size; (void)ws_size;
  const float* x = (const float*)d_in[0];
  float* out = (float*)d_out;

  char* ws = (char*)d_ws;
  unsigned short* xfbf = (unsigned short*)ws; ws += (size_t)N_PTS * C_DIM * 2; // 16 MB
  unsigned short* cbf  = (unsigned short*)ws; ws += (size_t)K_CTR * C_DIM * 2; // 0.5 MB
  float* ema    = (float*)ws; ws += (size_t)K_CTR * C_DIM * 4;                 // 1 MB
  float* sums   = (float*)ws; ws += (size_t)K_CTR * C_DIM * 4;                 // 1 MB
  float* counts = (float*)ws; ws += (size_t)K_CTR * 4;   // contiguous after sums
  float* c_sq   = (float*)ws; ws += (size_t)K_CTR * 4;
  int*   labels = (int*)ws;

  prep_bf16<<<(B_DIM * C_DIM * T_DIM) / 256, 256, 0, stream>>>(x, xfbf);
  init_centers<<<K_CTR, C_DIM, 0, stream>>>(x, ema);
  update_pack<<<K_CTR, C_DIM, 0, stream>>>(ema, sums, counts, cbf, c_sq, 0);

  for (int it = 0; it < ITERS; ++it) {
    assign_wmma<<<N_PTS / 256, 256, 0, stream>>>(xfbf, cbf, c_sq, labels, nullptr);
    zero_buf<<<(K_CTR * C_DIM + K_CTR + 255) / 256, 256, 0, stream>>>(
        sums, K_CTR * C_DIM + K_CTR);
    accumulate<<<N_PTS, C_DIM, 0, stream>>>(x, labels, sums, counts);
    update_pack<<<K_CTR, C_DIM, 0, stream>>>(ema, sums, counts, cbf, c_sq, 1);
  }
  // final labels against final ema centers, written as float to d_out
  assign_wmma<<<N_PTS / 256, 256, 0, stream>>>(xfbf, cbf, c_sq, labels, out);
}
